// scSimGCLCore_27487790694613
// MI455X (gfx1250) — matrix operand order repaired and checked by reading, not compiled, not run on hardware
//
#include <hip/hip_runtime.h>
#include <hip/hip_bf16.h>
#include <cstdint>

typedef __attribute__((ext_vector_type(16))) _Float16 v16h;
typedef __attribute__((ext_vector_type(8)))  _Float16 v8h;
typedef __attribute__((ext_vector_type(8)))  float    v8f;

#define NROWS 8192
#define DDIM  64
#define LDS_PITCH 72          // 64 + 8 halfs pad -> conflict-free ds_load_b128
#define TAU_INV 2.0f          // 1 / tau, tau = 0.5
#define E2 7.3890560989306495f // exp(1/tau) = intra diagonal
#define EPSV 1e-8f

// ---------------- fragment loaders (ISA 7.12.2 f16 layouts) ----------------

// A 16x32 f16: lane m = lane&15, half h = lane>>4.
//   elements 0..7  hold K = h*8 + 0..7      (+32*ks)
//   elements 8..15 hold K = 16 + h*8 + 0..7 (+32*ks)
__device__ __forceinline__ v16h load_a_frag(const _Float16* rowBase, int h, int ks) {
  const _Float16* p = rowBase + ks * 32 + h * 8;
  v8h lo = *(const v8h*)p;
  v8h hi = *(const v8h*)(p + 16);
  return __builtin_shufflevector(lo, hi, 0,1,2,3,4,5,6,7,8,9,10,11,12,13,14,15);
}

// B 32x16 f16: lane n = lane&15 holds column n; half h = lane>>4.
//   elements 0..7 hold K = h*8 + 0..7, elements 8..15 hold K = 16 + h*8 + 0..7 (+32*ks)
__device__ __forceinline__ v16h load_b_frag(const _Float16* ldsRow, int h, int ks) {
  const _Float16* p = ldsRow + ks * 32 + h * 8;
  v8h lo = *(const v8h*)p;
  v8h hi = *(const v8h*)(p + 16);
  return __builtin_shufflevector(lo, hi, 0,1,2,3,4,5,6,7,8,9,10,11,12,13,14,15);
}

// ---------------- phase 0: normalize rows, emit f16 + inter diagonal -------

__global__ __launch_bounds__(256) void normalize_kernel(
    const float* __restrict__ z, const float* __restrict__ za,
    _Float16* __restrict__ zn, _Float16* __restrict__ zan,
    float* __restrict__ diagC)
{
  const int wave = threadIdx.x >> 5;
  const int lane = threadIdx.x & 31;
  const int row  = blockIdx.x * 8 + wave;

  const float2 a = ((const float2*)(z  + (size_t)row * DDIM))[lane];
  const float2 b = ((const float2*)(za + (size_t)row * DDIM))[lane];

  float sz = a.x * a.x + a.y * a.y;
  float sa = b.x * b.x + b.y * b.y;
  float dt = a.x * b.x + a.y * b.y;
  #pragma unroll
  for (int m = 16; m >= 1; m >>= 1) {
    sz += __shfl_xor(sz, m, 32);
    sa += __shfl_xor(sa, m, 32);
    dt += __shfl_xor(dt, m, 32);
  }
  const float iz = 1.0f / fmaxf(sqrtf(sz), 1e-12f);
  const float ia = 1.0f / fmaxf(sqrtf(sa), 1e-12f);

  zn [(size_t)row * DDIM + 2 * lane    ] = (_Float16)(a.x * iz);
  zn [(size_t)row * DDIM + 2 * lane + 1] = (_Float16)(a.y * iz);
  zan[(size_t)row * DDIM + 2 * lane    ] = (_Float16)(b.x * ia);
  zan[(size_t)row * DDIM + 2 * lane + 1] = (_Float16)(b.y * ia);
  if (lane == 0) diagC[row] = __expf(TAU_INV * dt * iz * ia);
}

// ---------------- zero workspace accumulators -------------------------------

__global__ void zero_kernel(float* __restrict__ p, int nelem) {
  int i = blockIdx.x * blockDim.x + threadIdx.x;
  if (i < nelem) p[i] = 0.0f;
}

// ---------------- main fused pass -------------------------------------------
// Per pass: rows from zr.  intra gram = zr @ zr^T, inter gram = zr @ zc^T.
// Accumulates per-row: sum(exp), sum(exp*adj) for both grams, and sum(adj).

__global__ __launch_bounds__(256) void pass_kernel(
    const _Float16* __restrict__ zr, const _Float16* __restrict__ zc,
    const float* __restrict__ adj,
    float* __restrict__ sumI, float* __restrict__ posI,
    float* __restrict__ sumX, float* __restrict__ posX,
    float* __restrict__ cnt)
{
  __shared__ _Float16 ldsI[128 * LDS_PITCH];
  __shared__ _Float16 ldsX[128 * LDS_PITCH];

  const int tid  = threadIdx.x;
  const int wave = tid >> 5;
  const int lane = tid & 31;
  const int n    = lane & 15;
  const int h    = lane >> 4;

  const int rbase = blockIdx.x * 128;
  const int arow  = rbase + wave * 16 + n;

  const v16h a0 = load_a_frag(zr + (size_t)arow * DDIM, h, 0);
  const v16h a1 = load_a_frag(zr + (size_t)arow * DDIM, h, 1);

  // accumulator element e corresponds to global row rbase + wave*16 + h*8 + e
  float rsI[8], rpI[8], rsX[8], rpX[8], rc[8];
  #pragma unroll
  for (int e = 0; e < 8; ++e) { rsI[e] = rpI[e] = rsX[e] = rpX[e] = rc[e] = 0.0f; }

  const float* adjRow[8];
  #pragma unroll
  for (int e = 0; e < 8; ++e)
    adjRow[e] = adj + (size_t)(rbase + wave * 16 + h * 8 + e) * NROWS;

  for (int t = 0; t < 8; ++t) {
    const int jb = (blockIdx.y * 8 + t) * 128;

    __syncthreads();
    { // cooperative stage of the 128x64 column blocks (both matrices) into LDS
      const int r = tid >> 1, s = tid & 1;
      const _Float16* gi = zr + (size_t)(jb + r) * DDIM + s * 32;
      const _Float16* gx = zc + (size_t)(jb + r) * DDIM + s * 32;
      _Float16* di = ldsI + r * LDS_PITCH + s * 32;
      _Float16* dx = ldsX + r * LDS_PITCH + s * 32;
      #pragma unroll
      for (int u = 0; u < 4; ++u) {
        ((v8h*)di)[u] = ((const v8h*)gi)[u];
        ((v8h*)dx)[u] = ((const v8h*)gx)[u];
      }
    }
    __syncthreads();

    #pragma unroll
    for (int c = 0; c < 8; ++c) {
      const _Float16* rowI = ldsI + (c * 16 + n) * LDS_PITCH;
      const _Float16* rowX = ldsX + (c * 16 + n) * LDS_PITCH;
      const v16h bI0 = load_b_frag(rowI, h, 0);
      const v16h bI1 = load_b_frag(rowI, h, 1);
      const v16h bX0 = load_b_frag(rowX, h, 0);
      const v16h bX1 = load_b_frag(rowX, h, 1);

      v8f accI = {}; v8f accX = {};
      accI = __builtin_amdgcn_wmma_f32_16x16x32_f16(false, a0, false, bI0, (short)0, accI, false, false);
      accI = __builtin_amdgcn_wmma_f32_16x16x32_f16(false, a1, false, bI1, (short)0, accI, false, false);
      accX = __builtin_amdgcn_wmma_f32_16x16x32_f16(false, a0, false, bX0, (short)0, accX, false, false);
      accX = __builtin_amdgcn_wmma_f32_16x16x32_f16(false, a1, false, bX1, (short)0, accX, false, false);

      const int col = jb + c * 16 + n;
      #pragma unroll
      for (int e = 0; e < 8; ++e) {
        const float av = adjRow[e][col];
        const float eI = __expf(TAU_INV * accI[e]);
        const float eX = __expf(TAU_INV * accX[e]);
        rsI[e] += eI;
        rpI[e]  = fmaf(eI, av, rpI[e]);
        rsX[e] += eX;
        rpX[e]  = fmaf(eX, av, rpX[e]);
        rc[e]  += av;
      }
    }
  }

  // reduce across the 16 lanes (column index n) sharing each (h, e) row
  #pragma unroll
  for (int e = 0; e < 8; ++e) {
    #pragma unroll
    for (int m = 8; m >= 1; m >>= 1) {
      rsI[e] += __shfl_xor(rsI[e], m, 32);
      rpI[e] += __shfl_xor(rpI[e], m, 32);
      rsX[e] += __shfl_xor(rsX[e], m, 32);
      rpX[e] += __shfl_xor(rpX[e], m, 32);
      rc[e]  += __shfl_xor(rc[e],  m, 32);
    }
  }
  if (n == 0) {
    #pragma unroll
    for (int e = 0; e < 8; ++e) {
      const int row = rbase + wave * 16 + h * 8 + e;
      unsafeAtomicAdd(&sumI[row], rsI[e]);
      unsafeAtomicAdd(&posI[row], rpI[e]);
      unsafeAtomicAdd(&sumX[row], rsX[e]);
      unsafeAtomicAdd(&posX[row], rpX[e]);
      unsafeAtomicAdd(&cnt[row],  rc[e]);
    }
  }
}

// ---------------- finalize: per-row loss, mean, combine ---------------------

__global__ __launch_bounds__(256) void finalize_kernel(
    const float* __restrict__ acc, const float* __restrict__ diagC,
    float* __restrict__ out)
{
  const float* sumI1 = acc + 0 * NROWS;
  const float* posI1 = acc + 1 * NROWS;
  const float* sumX1 = acc + 2 * NROWS;
  const float* posX1 = acc + 3 * NROWS;
  const float* cnt1  = acc + 4 * NROWS;
  const float* sumI2 = acc + 5 * NROWS;
  const float* posI2 = acc + 6 * NROWS;
  const float* sumX2 = acc + 7 * NROWS;
  const float* posX2 = acc + 8 * NROWS;
  const float* cnt2  = acc + 9 * NROWS;

  __shared__ float red[256];
  float s = 0.0f;
  for (int i = threadIdx.x; i < NROWS; i += 256) {
    const float d  = diagC[i];
    const float p1 = d + posI1[i] + posX1[i];
    const float q1 = sumI1[i] + sumX1[i] - E2;
    const float l1 = logf(p1 / (q1 + EPSV) + EPSV) / (2.0f * cnt1[i] + 1.0f);
    const float p2 = d + posI2[i] + posX2[i];
    const float q2 = sumI2[i] + sumX2[i] - E2;
    const float l2 = logf(p2 / (q2 + EPSV) + EPSV) / (2.0f * cnt2[i] + 1.0f);
    s += l1 + l2;
  }
  red[threadIdx.x] = s;
  __syncthreads();
  for (int st = 128; st > 0; st >>= 1) {
    if (threadIdx.x < st) red[threadIdx.x] += red[threadIdx.x + st];
    __syncthreads();
  }
  if (threadIdx.x == 0) out[0] = -0.5f * red[0] / (float)NROWS;
}

// ---------------- launch ----------------------------------------------------

extern "C" void kernel_launch(void* const* d_in, const int* in_sizes, int n_in,
                              void* d_out, int out_size, void* d_ws, size_t ws_size,
                              hipStream_t stream) {
  (void)in_sizes; (void)n_in; (void)out_size; (void)ws_size;
  const float* z    = (const float*)d_in[0];
  const float* za   = (const float*)d_in[1];
  const float* adj  = (const float*)d_in[2];
  const float* adjA = (const float*)d_in[3];

  char* ws = (char*)d_ws;
  _Float16* zn   = (_Float16*)ws;                       // N*64 halfs = 1 MB
  _Float16* zan  = (_Float16*)(ws + 1048576);           // 1 MB
  float*    diagC = (float*)(ws + 2097152);             // N floats
  float*    acc   = (float*)(ws + 2097152 + 32768);     // 10*N floats

  zero_kernel<<<(10 * NROWS + 255) / 256, 256, 0, stream>>>(acc, 10 * NROWS);
  normalize_kernel<<<NROWS / 8, 256, 0, stream>>>(z, za, zn, zan, diagC);

  dim3 grid(NROWS / 128, 8);
  pass_kernel<<<grid, 256, 0, stream>>>(zn, zan, adj,
      acc + 0 * NROWS, acc + 1 * NROWS, acc + 2 * NROWS, acc + 3 * NROWS, acc + 4 * NROWS);
  pass_kernel<<<grid, 256, 0, stream>>>(zan, zn, adjA,
      acc + 5 * NROWS, acc + 6 * NROWS, acc + 7 * NROWS, acc + 8 * NROWS, acc + 9 * NROWS);

  finalize_kernel<<<1, 256, 0, stream>>>(acc, diagC, (float*)d_out);
}
